// PoolingAttention_37512244363656
// MI455X (gfx1250) — matrix operand order repaired and checked by reading, not compiled, not run on hardware
//
#include <hip/hip_runtime.h>
#include <hip/hip_bf16.h>

typedef __bf16 bf16_t;
typedef bf16_t v16bf __attribute__((ext_vector_type(16)));
typedef bf16_t v8bf  __attribute__((ext_vector_type(8)));
typedef float  v8f   __attribute__((ext_vector_type(8)));

#define B_   16
#define N_   3136
#define C_   512
#define H_   8
#define D_   64
#define HH_  56
#define AG_  144
#define AGP_ 160
#define PS_  12

// ---------------------------------------------------------------- WMMA helpers

__device__ __forceinline__ v8f wmma_bf16(v16bf a, v16bf b, v8f c) {
  return __builtin_amdgcn_wmma_f32_16x16x32_bf16(false, a, false, b, (short)0, c,
                                                 false, false);
}

// A fragment 16x32 (MxK): rowbase points at row m0 of a row-major [M][ldk] bf16
// matrix. Lane l<16 holds M=l, K = kc+[0..15]; lanes>=16 hold
// K = kc+16..kc+31  (two 16B loads per lane).
__device__ __forceinline__ v16bf load_a_frag(const bf16_t* rowbase, int ldk,
                                             int lane, int kc) {
  const bf16_t* p = rowbase + (size_t)(lane & 15) * ldk + kc + ((lane >> 4) << 4);
  v8bf lo = *(const v8bf*)p;
  v8bf hi = *(const v8bf*)(p + 8);
  return __builtin_shufflevector(lo, hi, 0, 1, 2, 3, 4, 5, 6, 7, 8, 9, 10, 11,
                                 12, 13, 14, 15);
}

// B fragment 32x16 (KxN) from a "column-contiguous" layout: colbase[n][k] rows
// are output-columns with contiguous K, row stride ldk. One 32B load per lane.
__device__ __forceinline__ v16bf load_b_frag(const bf16_t* colbase, size_t ldk,
                                             int lane, int n0, int kc) {
  const bf16_t* p =
      colbase + (size_t)(n0 + (lane & 15)) * ldk + kc + ((lane >> 4) << 4);
  return *(const v16bf*)p;
}

// C-fragment rows live across a 16-lane half (xor masks 1..8 stay in-half).
__device__ __forceinline__ float redmax16(float v) {
#pragma unroll
  for (int m = 1; m < 16; m <<= 1) v = fmaxf(v, __shfl_xor(v, m, 32));
  return v;
}
__device__ __forceinline__ float redsum16(float v) {
#pragma unroll
  for (int m = 1; m < 16; m <<= 1) v += __shfl_xor(v, m, 32);
  return v;
}

// ---------------------------------------------------------------- fp32 -> bf16

__global__ void cvt_kernel(const float* __restrict__ s, bf16_t* __restrict__ d,
                           unsigned long long n) {
  unsigned long long i =
      (unsigned long long)blockIdx.x * blockDim.x + threadIdx.x;
  if (i < n) d[i] = (bf16_t)s[i];
}

// ----------------------------------------------- qkv GEMM: [50176x512]x[512x1536]
// Wave tile 16x128 (8 C-fragments), software-pipelined: next k-chunk's 18 loads
// issue before the current chunk's 8 WMMAs so the wait tolerates in-flight loads.
// Scattered epilogue routes q->qh[b][h][n][d], k->kh[b][h][n][d],
// v->vT[b][c][n] (== NCHW image for the depthwise conv).

__global__ void gemm_qkv_kernel(const bf16_t* __restrict__ xb,
                                const bf16_t* __restrict__ wb,
                                bf16_t* __restrict__ qh, bf16_t* __restrict__ kh,
                                bf16_t* __restrict__ vT) {
  int wv = threadIdx.x >> 5, lane = threadIdx.x & 31;
  int gw = blockIdx.x * 8 + wv;
  int mt = gw / 12, nt = gw % 12;
  int m0 = mt * 16, n0 = nt * 128;
  int half = lane >> 4, col = lane & 15;

  v8f acc[8];
#pragma unroll
  for (int j = 0; j < 8; ++j)
#pragma unroll
    for (int r = 0; r < 8; ++r) acc[j][r] = 0.f;

  const bf16_t* arow = xb + (size_t)m0 * C_;

  v16bf a_c = load_a_frag(arow, C_, lane, 0);
  v16bf b_c[8];
#pragma unroll
  for (int j = 0; j < 8; ++j) b_c[j] = load_b_frag(wb, C_, lane, n0 + j * 16, 0);

#pragma unroll 2
  for (int kc = 32; kc < C_; kc += 32) {
    v16bf a_n = load_a_frag(arow, C_, lane, kc);
    v16bf b_n[8];
#pragma unroll
    for (int j = 0; j < 8; ++j)
      b_n[j] = load_b_frag(wb, C_, lane, n0 + j * 16, kc);
#pragma unroll
    for (int j = 0; j < 8; ++j) acc[j] = wmma_bf16(a_c, b_c[j], acc[j]);
    a_c = a_n;
#pragma unroll
    for (int j = 0; j < 8; ++j) b_c[j] = b_n[j];
  }
#pragma unroll
  for (int j = 0; j < 8; ++j) acc[j] = wmma_bf16(a_c, b_c[j], acc[j]);

  int mbb[8], mnn[8];
#pragma unroll
  for (int r = 0; r < 8; ++r) {
    int m = m0 + r + half * 8;
    mbb[r] = m / N_;
    mnn[r] = m % N_;
  }
#pragma unroll
  for (int j = 0; j < 8; ++j) {
    int o = n0 + j * 16 + col;
    int which = o >> 9, c = o & 511;
    int h = c >> 6, d = c & 63;
#pragma unroll
    for (int r = 0; r < 8; ++r) {
      bf16_t v = (bf16_t)acc[j][r];
      if (which == 0)
        qh[(((size_t)mbb[r] * H_ + h) * N_ + mnn[r]) * D_ + d] = v;
      else if (which == 1)
        kh[(((size_t)mbb[r] * H_ + h) * N_ + mnn[r]) * D_ + d] = v;
      else
        vT[((size_t)mbb[r] * C_ + c) * N_ + mnn[r]] = v;
    }
  }
}

// ------------------------------------------- adaptive 12x12 pool of q -> agents
// Scale d^-0.5 = 0.125 folded in (serves both score products). Also zero-fills
// the a=[144,160) padding of agent_h and agent_vT.

__global__ void pool_kernel(const bf16_t* __restrict__ qh,
                            bf16_t* __restrict__ agent,
                            bf16_t* __restrict__ agvT) {
  int a = blockIdx.x % AGP_, b = blockIdx.x / AGP_;
  int c = threadIdx.x;
  int h = c >> 6, d = c & 63;
  if (a >= AG_) {
    agent[(((size_t)b * H_ + h) * AGP_ + a) * D_ + d] = (bf16_t)0.f;
    agvT[(((size_t)b * H_ + h) * D_ + d) * AGP_ + a] = (bf16_t)0.f;
    return;
  }
  int p = a / PS_, q = a % PS_;
  int ys = (p * HH_) / PS_, ye = ((p + 1) * HH_ + PS_ - 1) / PS_;
  int xs = (q * HH_) / PS_, xe = ((q + 1) * HH_ + PS_ - 1) / PS_;
  const bf16_t* base = qh + (((size_t)b * H_ + h) * N_) * D_ + d;
  float s = 0.f;
  for (int y = ys; y < ye; ++y)
    for (int x = xs; x < xe; ++x) s += (float)base[(size_t)(y * HH_ + x) * D_];
  float val = s / (float)((ye - ys) * (xe - xs)) * 0.125f;
  agent[(((size_t)b * H_ + h) * AGP_ + a) * D_ + d] = (bf16_t)val;
}

// ------------------------------------------------- depthwise 3x3 conv on vT(NCHW)

__global__ void dwc_kernel(const bf16_t* __restrict__ vT,
                           const float* __restrict__ w,
                           const float* __restrict__ bias,
                           bf16_t* __restrict__ dw) {
  size_t idx = (size_t)blockIdx.x * 256 + threadIdx.x;  // == b*C*N + c*N + n
  int n = (int)(idx % N_);
  size_t bc = idx / N_;
  int c = (int)(bc % C_);
  int y = n / HH_, x = n % HH_;
  const bf16_t* in = vT + bc * N_;
  const float* wc = w + c * 9;
  float s = bias[c];
#pragma unroll
  for (int ky = 0; ky < 3; ++ky) {
    int yy = y + ky - 1;
    if (yy < 0 || yy >= HH_) continue;
#pragma unroll
    for (int kx = 0; kx < 3; ++kx) {
      int xx = x + kx - 1;
      if (xx < 0 || xx >= HH_) continue;
      s += (float)in[yy * HH_ + xx] * wc[ky * 3 + kx];
    }
  }
  dw[idx] = (bf16_t)s;
}

// ------------------------- stage 1: agent_v = softmax(agent@K^T) @ V  (flash-style)
// One block per (b,h); 9 waves, each owns 16 agents and streams 3136 keys in
// 32-key tiles with online softmax; result written transposed agent_vT[d][a].
// K and V fragments for a tile load up front: V load latency hides under the
// softmax VALU work.

__global__ void stage1_kernel(const bf16_t* __restrict__ agent,
                              const bf16_t* __restrict__ kh,
                              const bf16_t* __restrict__ vT,
                              bf16_t* __restrict__ agvT) {
  __shared__ bf16_t lds[9 * 16 * 32];
  int bh = blockIdx.x;
  int wv = threadIdx.x >> 5, lane = threadIdx.x & 31;
  int half = lane >> 4, col = lane & 15;
  int a0 = wv * 16;
  int b = bh >> 3, h = bh & 7;

  const bf16_t* agrow = agent + ((size_t)bh * AGP_ + a0) * D_;
  v16bf aa0 = load_a_frag(agrow, D_, lane, 0);
  v16bf aa1 = load_a_frag(agrow, D_, lane, 32);
  const bf16_t* khb = kh + (size_t)bh * N_ * D_;
  const bf16_t* vtb = vT + ((size_t)b * C_ + h * D_) * N_;
  bf16_t* lp = lds + wv * 512;  // wave-private 16x32 P tile

  float mrun[8], srun[8];
  v8f acc[4];
#pragma unroll
  for (int r = 0; r < 8; ++r) {
    mrun[r] = -1e30f;
    srun[r] = 0.f;
  }
#pragma unroll
  for (int t = 0; t < 4; ++t)
#pragma unroll
    for (int r = 0; r < 8; ++r) acc[t][r] = 0.f;

  for (int j0 = 0; j0 < N_; j0 += 32) {
    // All global fragment loads for this tile issue first.
    v16bf kf0 = load_b_frag(khb, D_, lane, j0, 0);
    v16bf kf1 = load_b_frag(khb, D_, lane, j0, 32);
    v16bf kf2 = load_b_frag(khb, D_, lane, j0 + 16, 0);
    v16bf kf3 = load_b_frag(khb, D_, lane, j0 + 16, 32);
    v16bf vfr[4];
#pragma unroll
    for (int t = 0; t < 4; ++t)
      vfr[t] = *(const v16bf*)(vtb + (size_t)(t * 16 + col) * N_ + j0 + half * 16);

    v8f S0, S1;
#pragma unroll
    for (int r = 0; r < 8; ++r) {
      S0[r] = 0.f;
      S1[r] = 0.f;
    }
    S0 = wmma_bf16(aa0, kf0, S0);
    S0 = wmma_bf16(aa1, kf1, S0);
    S1 = wmma_bf16(aa0, kf2, S1);
    S1 = wmma_bf16(aa1, kf3, S1);

    float corr[8], p0[8], p1[8];
#pragma unroll
    for (int r = 0; r < 8; ++r) {
      float mx = redmax16(fmaxf(S0[r], S1[r]));
      float mnew = fmaxf(mrun[r], mx);
      corr[r] = __expf(mrun[r] - mnew);
      p0[r] = __expf(S0[r] - mnew);
      p1[r] = __expf(S1[r] - mnew);
      float rs = redsum16(p0[r] + p1[r]);
      srun[r] = srun[r] * corr[r] + rs;
      mrun[r] = mnew;
    }
#pragma unroll
    for (int t = 0; t < 4; ++t)
#pragma unroll
      for (int r = 0; r < 8; ++r) acc[t][r] *= corr[r];

    // C-layout -> A-layout via wave-private LDS (DS ops in-order per wave)
#pragma unroll
    for (int r = 0; r < 8; ++r) {
      lp[(r + half * 8) * 32 + col] = (bf16_t)p0[r];
      lp[(r + half * 8) * 32 + 16 + col] = (bf16_t)p1[r];
    }
    v16bf pa = load_a_frag(lp, 32, lane, 0);
#pragma unroll
    for (int t = 0; t < 4; ++t) acc[t] = wmma_bf16(pa, vfr[t], acc[t]);
  }

#pragma unroll
  for (int r = 0; r < 8; ++r) srun[r] = 1.f / srun[r];
#pragma unroll
  for (int t = 0; t < 4; ++t)
#pragma unroll
    for (int r = 0; r < 8; ++r) {
      int a = a0 + r + half * 8;
      int d = t * 16 + col;
      agvT[((size_t)bh * D_ + d) * AGP_ + a] = (bf16_t)(acc[t][r] * srun[r]);
    }
}

// -------------------- stage 2: out = softmax(q@agent^T) @ agent_v  (+ dwc fuse)
// 144 agents = exactly 9 score fragments; K padded to 160 (zero P / zero V) so
// P@V runs as 5 clean K=32 WMMA chunks. Epilogue adds conv residual, emits bf16.

__global__ void stage2_kernel(const bf16_t* __restrict__ qh,
                              const bf16_t* __restrict__ agent,
                              const bf16_t* __restrict__ agvT,
                              const bf16_t* __restrict__ dw,
                              bf16_t* __restrict__ attn) {
  __shared__ bf16_t lds[8 * 16 * 160];
  int bh = blockIdx.x / 25, grp = blockIdx.x % 25;
  int wv = threadIdx.x >> 5, lane = threadIdx.x & 31;
  int qt = grp * 8 + wv;
  if (qt >= 196) return;  // wave-uniform exit, no barriers used
  int q0 = qt * 16;
  int half = lane >> 4, col = lane & 15;
  int b = bh >> 3, h = bh & 7;

  const bf16_t* qrow = qh + ((size_t)bh * N_ + q0) * D_;
  v16bf aq0 = load_a_frag(qrow, D_, lane, 0);
  v16bf aq1 = load_a_frag(qrow, D_, lane, 32);
  const bf16_t* agb = agent + (size_t)bh * AGP_ * D_;

  // Batch all 18 score-operand loads, then run the 18 WMMAs back-to-back.
  v16bf bf0[9], bf1[9];
#pragma unroll
  for (int nt = 0; nt < 9; ++nt) {
    bf0[nt] = load_b_frag(agb, D_, lane, nt * 16, 0);
    bf1[nt] = load_b_frag(agb, D_, lane, nt * 16, 32);
  }
  v8f S[9];
#pragma unroll
  for (int nt = 0; nt < 9; ++nt) {
#pragma unroll
    for (int r = 0; r < 8; ++r) S[nt][r] = 0.f;
    S[nt] = wmma_bf16(aq0, bf0[nt], S[nt]);
    S[nt] = wmma_bf16(aq1, bf1[nt], S[nt]);
  }
  float mx[8], sum[8];
#pragma unroll
  for (int r = 0; r < 8; ++r) {
    float m = S[0][r];
#pragma unroll
    for (int nt = 1; nt < 9; ++nt) m = fmaxf(m, S[nt][r]);
    mx[r] = redmax16(m);
  }
#pragma unroll
  for (int r = 0; r < 8; ++r) {
    float rs = 0.f;
#pragma unroll
    for (int nt = 0; nt < 9; ++nt) {
      float p = __expf(S[nt][r] - mx[r]);
      S[nt][r] = p;
      rs += p;
    }
    sum[r] = 1.f / redsum16(rs);
  }
  bf16_t* lp = lds + wv * 16 * 160;  // wave-private 16x160 P tile
#pragma unroll
  for (int r = 0; r < 8; ++r) {
    int row = (r + half * 8) * 160;
#pragma unroll
    for (int nt = 0; nt < 9; ++nt) lp[row + nt * 16 + col] = (bf16_t)S[nt][r];
    lp[row + 144 + col] = (bf16_t)0.f;  // padding columns
  }
  v8f acc[4];
#pragma unroll
  for (int t = 0; t < 4; ++t)
#pragma unroll
    for (int r = 0; r < 8; ++r) acc[t][r] = 0.f;
#pragma unroll
  for (int kc = 0; kc < 5; ++kc) {
    v16bf vb[4];
#pragma unroll
    for (int t = 0; t < 4; ++t)
      vb[t] = *(const v16bf*)(agvT + ((size_t)bh * D_ + t * 16 + col) * AGP_ +
                              kc * 32 + half * 16);
    v16bf pa = load_a_frag(lp, 160, lane, kc * 32);
#pragma unroll
    for (int t = 0; t < 4; ++t) acc[t] = wmma_bf16(pa, vb[t], acc[t]);
  }
#pragma unroll
  for (int t = 0; t < 4; ++t) {
    int c = h * D_ + t * 16 + col;
#pragma unroll
    for (int r = 0; r < 8; ++r) {
      int n = q0 + r + half * 8;
      float dv = (float)dw[((size_t)b * C_ + c) * N_ + n];
      attn[((size_t)b * N_ + n) * C_ + c] = (bf16_t)(acc[t][r] * sum[r] + dv);
    }
  }
}

// ------------------------------------------- proj GEMM: [50176x512]x[512x512]+b
// Same double-buffered pipeline as the qkv GEMM.

__global__ void gemm_proj_kernel(const bf16_t* __restrict__ attn,
                                 const bf16_t* __restrict__ pw,
                                 const float* __restrict__ pb,
                                 float* __restrict__ out) {
  int wv = threadIdx.x >> 5, lane = threadIdx.x & 31;
  int gw = blockIdx.x * 8 + wv;
  int mt = gw >> 2, nt = gw & 3;
  int m0 = mt * 16, n0 = nt * 128;
  int half = lane >> 4, col = lane & 15;

  v8f acc[8];
#pragma unroll
  for (int j = 0; j < 8; ++j)
#pragma unroll
    for (int r = 0; r < 8; ++r) acc[j][r] = 0.f;

  const bf16_t* arow = attn + (size_t)m0 * C_;

  v16bf a_c = load_a_frag(arow, C_, lane, 0);
  v16bf b_c[8];
#pragma unroll
  for (int j = 0; j < 8; ++j) b_c[j] = load_b_frag(pw, C_, lane, n0 + j * 16, 0);

#pragma unroll 2
  for (int kc = 32; kc < C_; kc += 32) {
    v16bf a_n = load_a_frag(arow, C_, lane, kc);
    v16bf b_n[8];
#pragma unroll
    for (int j = 0; j < 8; ++j)
      b_n[j] = load_b_frag(pw, C_, lane, n0 + j * 16, kc);
#pragma unroll
    for (int j = 0; j < 8; ++j) acc[j] = wmma_bf16(a_c, b_c[j], acc[j]);
    a_c = a_n;
#pragma unroll
    for (int j = 0; j < 8; ++j) b_c[j] = b_n[j];
  }
#pragma unroll
  for (int j = 0; j < 8; ++j) acc[j] = wmma_bf16(a_c, b_c[j], acc[j]);

#pragma unroll
  for (int j = 0; j < 8; ++j) {
    int o = n0 + j * 16 + col;
    float bias = pb[o];
#pragma unroll
    for (int r = 0; r < 8; ++r) {
      int m = m0 + r + half * 8;
      out[(size_t)m * C_ + o] = acc[j][r] + bias;  // coalesced b32 stores
    }
  }
}

// ---------------------------------------------------------------- launch

extern "C" void kernel_launch(void* const* d_in, const int* in_sizes, int n_in,
                              void* d_out, int out_size, void* d_ws,
                              size_t ws_size, hipStream_t stream) {
  (void)in_sizes; (void)n_in; (void)out_size; (void)ws_size;
  const float* x      = (const float*)d_in[0];
  const float* qkv_w  = (const float*)d_in[1];
  const float* proj_w = (const float*)d_in[2];
  const float* proj_b = (const float*)d_in[3];
  const float* dwc_w  = (const float*)d_in[4];
  const float* dwc_b  = (const float*)d_in[5];

  constexpr size_t nBNC  = (size_t)B_ * N_ * C_;        // 25,690,112
  constexpr size_t nQKVW = 3ull * C_ * C_;              // 786,432
  constexpr size_t nPW   = (size_t)C_ * C_;             // 262,144
  constexpr size_t nAG   = (size_t)B_ * H_ * AGP_ * D_; // 1,310,720

  char* ws = (char*)d_ws;
  bf16_t* xb    = (bf16_t*)ws; ws += nBNC * 2;
  bf16_t* wb    = (bf16_t*)ws; ws += nQKVW * 2;
  bf16_t* pwb   = (bf16_t*)ws; ws += nPW * 2;
  bf16_t* qhp   = (bf16_t*)ws; ws += nBNC * 2;   // q  [b][h][n][d]
  bf16_t* khp   = (bf16_t*)ws; ws += nBNC * 2;   // k  [b][h][n][d]
  bf16_t* vtp   = (bf16_t*)ws; ws += nBNC * 2;   // v  [b][c][n] (NCHW)
  bf16_t* agp   = (bf16_t*)ws; ws += nAG * 2;    // agents [b][h][160][64] (x0.125)
  bf16_t* avtp  = (bf16_t*)ws; ws += nAG * 2;    // agent_v^T [b][h][64][160]
  bf16_t* dwp   = (bf16_t*)ws; ws += nBNC * 2;   // conv residual [b][c][n]
  bf16_t* attnp = (bf16_t*)ws; ws += nBNC * 2;   // attn+dwc [b][n][c]

  cvt_kernel<<<(unsigned)((nBNC + 255) / 256), 256, 0, stream>>>(x, xb, nBNC);
  cvt_kernel<<<(unsigned)((nQKVW + 255) / 256), 256, 0, stream>>>(qkv_w, wb, nQKVW);
  cvt_kernel<<<(unsigned)((nPW + 255) / 256), 256, 0, stream>>>(proj_w, pwb, nPW);

  gemm_qkv_kernel<<<4704, 256, 0, stream>>>(xb, wb, qhp, khp, vtp);
  pool_kernel<<<B_ * AGP_, 512, 0, stream>>>(qhp, agp, avtp);
  dwc_kernel<<<100352, 256, 0, stream>>>(vtp, dwc_w, dwc_b, dwp);
  stage1_kernel<<<B_ * H_, 288, 0, stream>>>(agp, khp, vtp, avtp);
  stage2_kernel<<<B_ * H_ * 25, 256, 0, stream>>>(qhp, agp, avtp, dwp, attnp);
  gemm_proj_kernel<<<1568, 256, 0, stream>>>(attnp, pwb, proj_b, (float*)d_out);
}
